// WindowAttention_71004399338066
// MI455X (gfx1250) — matrix operand order repaired and checked by reading, not compiled, not run on hardware
//
#include <hip/hip_runtime.h>

// ---------------------------------------------------------------------------
// Fully-fused Swin window attention for gfx1250 (MI455X).
// One workgroup (384 threads = 12 wave32) per window; everything between the
// x-read and the out-write stays in the 320KB WGP LDS. Matrix math uses
// v_wmma_f32_16x16x32_f16 (f16 in, f32 accumulate).
// ---------------------------------------------------------------------------

typedef __attribute__((ext_vector_type(16))) _Float16 v16h;
typedef __attribute__((ext_vector_type(8)))  _Float16 v8h;
typedef __attribute__((ext_vector_type(8)))  float    v8f;

#define HEADS   12
#define NTOK    49
#define NPAD    64
#define DH      32
#define DIMC    384
#define QKV_DIM 1152
#define NWIN    4096
#define NMASK   64

// ---- LDS layout (units: halves / _Float16) --------------------------------
#define XAO_STRIDE 392                        // 384 + 8 pad (784B rows, 16B aligned)
#define XAO_OFF    0                          // [64][392] : x staging, reused as attn-out
#define QKV_OFF    (NPAD * XAO_STRIDE)        // 25088
#define QK_STRIDE  40                         // 32 + 8 pad per q/k row
#define VT_STRIDE  72                         // 64 + 8 pad per v^T row
#define K_SUB      (NPAD * QK_STRIDE)         // 2560
#define VT_SUB     (2 * NPAD * QK_STRIDE)     // 5120
#define HEAD_STRIDE (2 * NPAD * QK_STRIDE + DH * VT_STRIDE)  // 7424
#define P_OFF      (QKV_OFF + HEADS * HEAD_STRIDE)           // 114176
#define P_STRIDE   72
#define SMEM_HALVES (P_OFF + HEADS * 16 * P_STRIDE)          // 128000
#define SMEM_BYTES  (SMEM_HALVES * 2)                        // 256000 < 320KB

static __device__ __forceinline__ v16h mk16(v8h lo, v8h hi) {
  v16h r;
#pragma unroll
  for (int i = 0; i < 8; ++i) { r[i] = lo[i]; r[i + 8] = hi[i]; }
  return r;
}

// A fragment (16x32 f16) from a row-major source. ISA 7.12.2:
//   lanes 0-15 : M = lane,    K in {0..7, 16..23}  (+k0)
//   lanes 16-31: M = lane-16, K in {8..15, 24..31} (+k0)
static __device__ __forceinline__ v16h load_a(const _Float16* base, int stride,
                                              int row0, int k0, int lane) {
  const int m  = row0 + (lane & 15);
  const int kb = k0 + ((lane >> 4) << 3);
  const _Float16* p = base + m * stride + kb;
  return mk16(*(const v8h*)p, *(const v8h*)(p + 16));
}

// B fragment (32x16 f16). Source row n holds B column n contiguously
// (k rows / v^T rows / weight rows). ISA 7.12.4 analog (wave32):
//   lanes 0-15 : N = lane,    K in {0..7, 16..23}  (+k0)
//   lanes 16-31: N = lane-16, K in {8..15, 24..31} (+k0)
static __device__ __forceinline__ v16h load_b(const _Float16* base, int stride,
                                              int col0, int k0, int lane) {
  const int n  = col0 + (lane & 15);
  const int kb = k0 + ((lane >> 4) << 3);
  const _Float16* p = base + n * stride + kb;
  return mk16(*(const v8h*)p, *(const v8h*)(p + 16));
}

static __device__ __forceinline__ v8f wmma16(v16h a, v16h b, v8f c) {
  return __builtin_amdgcn_wmma_f32_16x16x32_f16(false, a, false, b,
                                                (short)0, c, false, false);
}

// ---------------------------------------------------------------------------
__global__ void convert_weights(const float* __restrict__ qkvw,
                                const float* __restrict__ outw,
                                _Float16* __restrict__ qkvw16,
                                _Float16* __restrict__ outw16) {
  const int i = blockIdx.x * 256 + threadIdx.x;
  if (i < QKV_DIM * DIMC) qkvw16[i] = (_Float16)qkvw[i];
  if (i < DIMC * DIMC)    outw16[i] = (_Float16)outw[i];
}

// ---------------------------------------------------------------------------
__global__ __launch_bounds__(384) void swin_window_attn(
    const float* __restrict__ x, const float* __restrict__ qkv_b,
    const float* __restrict__ rp_table, const float* __restrict__ out_b,
    const int* __restrict__ rp_index, const float* __restrict__ mask,
    const _Float16* __restrict__ qkvw16, const _Float16* __restrict__ outw16,
    float* __restrict__ out) {
  extern __shared__ _Float16 smem[];
  const int b    = blockIdx.x;
  const int tid  = threadIdx.x;
  const int w    = tid >> 5;      // wave id 0..11 (wave32)
  const int lane = tid & 31;
  const int nidx = lane & 15;
  const int hi16 = lane >> 4;
  const float scale = 0.1767766952966369f;  // 32^-0.5

  // ---- stage x_b (f32 -> f16) into LDS; zero rows 49..63 -------------------
  {
    const float* xb = x + (size_t)b * (NTOK * DIMC);
    for (int i = tid; i < NPAD * DIMC; i += 384) {
      const int r = i / DIMC, c = i - r * DIMC;
      const float v = (r < NTOK) ? xb[r * DIMC + c] : 0.f;
      smem[XAO_OFF + r * XAO_STRIDE + c] = (_Float16)v;
    }
  }
  __syncthreads();

  // ---- phase 1: qkv = x @ qkv_w^T + qkv_b, scattered into per-head LDS ----
  // 288 output tiles (4 row-tiles x 72 col-tiles); 24 per wave.
  for (int i = 0; i < 24; ++i) {
    const int t  = w * 24 + i;
    const int jc = t >> 2, rt = t & 3;
    __builtin_prefetch(qkvw16 + (size_t)(jc * 16 + nidx) * DIMC, 0, 3);
    v8f c = {};
#pragma unroll 4
    for (int kk = 0; kk < 12; ++kk) {
      const v16h a  = load_a(smem + XAO_OFF, XAO_STRIDE, rt * 16, kk * 32, lane);
      const v16h bf = load_b(qkvw16, DIMC, jc * 16, kk * 32, lane);
      c = wmma16(a, bf, c);
    }
    // epilogue: column o factors as o = h*96 + d*3 + s  (s: 0=q,1=k,2=v)
    const int o  = jc * 16 + nidx;
    const int s  = o % 3;
    const int d  = (o / 3) & 31;
    const int hh = o / 96;
    const float bias = qkv_b[o];
    const int base = (s == 2)
        ? (QKV_OFF + hh * HEAD_STRIDE + VT_SUB + d * VT_STRIDE)           // v^T[d][row]
        : (QKV_OFF + hh * HEAD_STRIDE + (s == 1 ? K_SUB : 0) + d);        // q/k[row][d]
#pragma unroll
    for (int r = 0; r < 8; ++r) {
      const int row = rt * 16 + hi16 * 8 + r;
      float v = c[r] + bias;
      if (s == 0) v *= scale;
      const int addr = base + ((s == 2) ? row : row * QK_STRIDE);
      smem[addr] = (_Float16)v;
    }
  }
  __syncthreads();

  // ---- phase 2: per-head attention (wave w owns head h = w) ---------------
  {
    const int h = w;
    _Float16* qh  = smem + QKV_OFF + h * HEAD_STRIDE;
    _Float16* kh  = qh + K_SUB;
    _Float16* vth = qh + VT_SUB;
    _Float16* pl  = smem + P_OFF + w * (16 * P_STRIDE);
    const int wid = b & (NMASK - 1);

    for (int rt = 0; rt < 4; ++rt) {
      const v16h aq = load_a(qh, QK_STRIDE, rt * 16, 0, lane);
      float sacc[4][8];
#pragma unroll
      for (int ct = 0; ct < 4; ++ct) {
        v8f c = {};
        c = wmma16(aq, load_b(kh, QK_STRIDE, ct * 16, 0, lane), c);
        const int mg = ct * 16 + nidx;
#pragma unroll
        for (int r = 0; r < 8; ++r) {
          const int ng = rt * 16 + hi16 * 8 + r;
          float v;
          if (ng < NTOK && mg < NTOK) {
            const int idx = rp_index[ng * NTOK + mg];
            v = c[r] + rp_table[idx * HEADS + h] +
                mask[wid * (NTOK * NTOK) + ng * NTOK + mg];
          } else {
            v = -1e30f;  // pad rows/cols
          }
          sacc[ct][r] = v;
        }
      }
      // row softmax: each row spans 16 lanes (within a half-wave) x 4 tiles
#pragma unroll
      for (int r = 0; r < 8; ++r) {
        float mx = fmaxf(fmaxf(sacc[0][r], sacc[1][r]),
                         fmaxf(sacc[2][r], sacc[3][r]));
#pragma unroll
        for (int off = 8; off; off >>= 1) mx = fmaxf(mx, __shfl_xor(mx, off, 16));
        float sum = 0.f;
#pragma unroll
        for (int ct = 0; ct < 4; ++ct) {
          const float e = __expf(sacc[ct][r] - mx);
          sacc[ct][r] = e;
          sum += e;
        }
#pragma unroll
        for (int off = 8; off; off >>= 1) sum += __shfl_xor(sum, off, 16);
        const float inv = 1.f / sum;
        const int rl = hi16 * 8 + r;  // C-layout row within tile
#pragma unroll
        for (int ct = 0; ct < 4; ++ct)
          pl[rl * P_STRIDE + ct * 16 + nidx] = (_Float16)(sacc[ct][r] * inv);
      }
      asm volatile("s_wait_dscnt 0x0" ::: "memory");  // P writes -> A reads

      v8f oc0 = {}, oc1 = {};
#pragma unroll
      for (int kk = 0; kk < 2; ++kk) {
        const v16h ap = load_a(pl, P_STRIDE, 0, kk * 32, lane);
        oc0 = wmma16(ap, load_b(vth, VT_STRIDE, 0,  kk * 32, lane), oc0);
        oc1 = wmma16(ap, load_b(vth, VT_STRIDE, 16, kk * 32, lane), oc1);
      }
#pragma unroll
      for (int r = 0; r < 8; ++r) {
        const int row = rt * 16 + hi16 * 8 + r;
        smem[XAO_OFF + row * XAO_STRIDE + h * 32 + nidx]      = (_Float16)oc0[r];
        smem[XAO_OFF + row * XAO_STRIDE + h * 32 + 16 + nidx] = (_Float16)oc1[r];
      }
      asm volatile("s_wait_dscnt 0x0" ::: "memory");  // before pl reuse
    }
  }
  __syncthreads();

  // ---- phase 3: out = ao @ out_w^T + out_b --------------------------------
  // 96 tiles (4 row-tiles x 24 col-tiles); 8 per wave.
  for (int i = 0; i < 8; ++i) {
    const int t  = w * 8 + i;
    const int jc = t >> 2, rt = t & 3;
    __builtin_prefetch(outw16 + (size_t)(jc * 16 + nidx) * DIMC, 0, 3);
    v8f c = {};
#pragma unroll 4
    for (int kk = 0; kk < 12; ++kk) {
      const v16h a  = load_a(smem + XAO_OFF, XAO_STRIDE, rt * 16, kk * 32, lane);
      const v16h bf = load_b(outw16, DIMC, jc * 16, kk * 32, lane);
      c = wmma16(a, bf, c);
    }
    const int o = jc * 16 + nidx;
    const float bo = out_b[o];
    float* ob = out + (size_t)b * (NTOK * DIMC);
#pragma unroll
    for (int r = 0; r < 8; ++r) {
      const int n = rt * 16 + hi16 * 8 + r;
      if (n < NTOK) ob[n * DIMC + o] = c[r] + bo;
    }
  }
}

// ---------------------------------------------------------------------------
extern "C" void kernel_launch(void* const* d_in, const int* in_sizes, int n_in,
                              void* d_out, int out_size, void* d_ws, size_t ws_size,
                              hipStream_t stream) {
  const float* x        = (const float*)d_in[0];
  const float* qkv_w    = (const float*)d_in[1];
  const float* qkv_b    = (const float*)d_in[2];
  const float* rp_table = (const float*)d_in[3];
  const float* out_w    = (const float*)d_in[4];
  const float* out_b    = (const float*)d_in[5];
  const int*   rp_index = (const int*)d_in[6];
  const float* mask     = (const float*)d_in[7];
  float* out = (float*)d_out;

  _Float16* qkvw16 = (_Float16*)d_ws;                 // 1152*384 halves
  _Float16* outw16 = qkvw16 + QKV_DIM * DIMC;         // +384*384 halves (~1.2MB total)

  convert_weights<<<(QKV_DIM * DIMC + 255) / 256, 256, 0, stream>>>(
      qkv_w, out_w, qkvw16, outw16);

  swin_window_attn<<<NWIN, 384, SMEM_BYTES, stream>>>(
      x, qkv_b, rp_table, out_b, rp_index, mask, qkvw16, outw16, out);
}
